// Atten_33509334843755
// MI455X (gfx1250) — matrix operand order repaired and checked by reading, not compiled, not run on hardware
//
#include <hip/hip_runtime.h>
#include <math.h>

// ---------------------------------------------------------------------------
// AttnBlock (GroupNorm + single-head spatial self-attention + residual)
// B=64, C=128, H=W=32 -> S=1024, 32 groups, scale = 1/sqrt(512)
// All GEMMs via v_wmma_f32_16x16x32_f16 (gfx1250 wave32 WMMA).
// ---------------------------------------------------------------------------

#define BATCH 64
#define CH    128
#define SEQ   1024
#define NGRP  32
#define EPS   1e-6f
#define ATT_SCALE 0.044194173824159216f

typedef __attribute__((ext_vector_type(16))) _Float16 v16h;
typedef __attribute__((ext_vector_type(8)))  float    v8f;

__device__ __forceinline__ v8f wmma16(v16h a, v16h b, v8f c) {
  return __builtin_amdgcn_wmma_f32_16x16x32_f16(
      /*neg_a=*/false, a, /*neg_b=*/false, b,
      /*c_mod=*/(short)0, c, /*reuse_a=*/false, /*reuse_b=*/false);
}

// A fragment, 16x32 f16, element (m,k) at p[m*ld + k]  (row-major)
// ISA layout: lanes 0-15 row=lane, K={0..7,16..23}; lanes 16-31 row=lane-16, K={8..15,24..31}
__device__ __forceinline__ v16h load_frag_a(const _Float16* p, int ld, int lane) {
  int row   = lane & 15;
  int kbase = (lane >> 4) * 8;
  const _Float16* r = p + row * ld + kbase;
  v16h a;
#pragma unroll
  for (int j = 0; j < 8; ++j) {
    a[j]     = r[j];        // K = kbase + j
    a[8 + j] = r[16 + j];   // K = kbase + 16 + j
  }
  return a;
}

// B fragment, 32x16 f16, element (k,n) at p[n*ld + k]  (k contiguous per lane)
// ISA layout: lanes 0-15 n=lane, K={0..7,16..23}; lanes 16-31 n=lane-16, K={8..15,24..31}
__device__ __forceinline__ v16h load_frag_b(const _Float16* p, int ld, int lane) {
  int n     = lane & 15;
  int kbase = (lane >> 4) * 8;
  const _Float16* r = p + n * ld + kbase;
  v16h b;
#pragma unroll
  for (int j = 0; j < 8; ++j) { b[j] = r[j]; b[8 + j] = r[16 + j]; }
  return b;
}

// ---------------------------------------------------------------------------
// Kernel 0: convert the four 128x128 f32 weights to f16 (kept [out][in])
// ---------------------------------------------------------------------------
__global__ void wconv_kernel(const float* __restrict__ wq, const float* __restrict__ wk,
                             const float* __restrict__ wv, const float* __restrict__ wo,
                             _Float16* __restrict__ w16) {
  int i = blockIdx.x * 256 + threadIdx.x;
  if (i < CH * CH) {
    w16[i]               = (_Float16)wq[i];
    w16[CH * CH + i]     = (_Float16)wk[i];
    w16[2 * CH * CH + i] = (_Float16)wv[i];
    w16[3 * CH * CH + i] = (_Float16)wo[i];
  }
}

// ---------------------------------------------------------------------------
// Kernel 1: GroupNorm -> t16 [B, S, C] f16.  One block per (b, group).
// ---------------------------------------------------------------------------
__global__ __launch_bounds__(256) void gn_kernel(const float* __restrict__ x,
                                                 const float* __restrict__ gw,
                                                 const float* __restrict__ gb,
                                                 _Float16* __restrict__ t16) {
  __shared__ float red_s[256];
  __shared__ float red_ss[256];
  int bg = blockIdx.x;
  int b  = bg >> 5;          // /NGRP
  int g  = bg & 31;
  const float* xp = x + ((size_t)b * CH + g * 4) * SEQ;  // 4 consecutive channels

  float s = 0.f, ss = 0.f;
  for (int i = threadIdx.x; i < 4 * SEQ; i += 256) {
    float v = xp[i];
    s += v; ss += v * v;
  }
  red_s[threadIdx.x] = s; red_ss[threadIdx.x] = ss;
  __syncthreads();
  for (int st = 128; st > 0; st >>= 1) {
    if (threadIdx.x < st) {
      red_s[threadIdx.x]  += red_s[threadIdx.x + st];
      red_ss[threadIdx.x] += red_ss[threadIdx.x + st];
    }
    __syncthreads();
  }
  float mean = red_s[0] * (1.f / 4096.f);
  float var  = red_ss[0] * (1.f / 4096.f) - mean * mean;
  float rstd = rsqrtf(var + EPS);

  for (int i = threadIdx.x; i < 4 * SEQ; i += 256) {
    int cl = i >> 10;             // 0..3
    int sp = i & (SEQ - 1);
    int c  = g * 4 + cl;
    float v = (xp[i] - mean) * rstd * gw[c] + gb[c];
    t16[((size_t)b * SEQ + sp) * CH + c] = (_Float16)v;
  }
}

// ---------------------------------------------------------------------------
// Kernel 2: QKV projections.  grid (S/64, 3, B), 8 waves/block, each wave a
// 16x64 output tile.  q is pre-scaled by 1/sqrt(512); v stored transposed.
// ---------------------------------------------------------------------------
__global__ __launch_bounds__(256) void qkv_kernel(const _Float16* __restrict__ t16,
                                                  const _Float16* __restrict__ w16,
                                                  const float* __restrict__ bq,
                                                  const float* __restrict__ bk,
                                                  const float* __restrict__ bv,
                                                  _Float16* __restrict__ q16,
                                                  _Float16* __restrict__ k16,
                                                  _Float16* __restrict__ vT16) {
  int which = blockIdx.y;                 // 0=q 1=k 2=v
  int b     = blockIdx.z;
  int wave  = threadIdx.x >> 5;
  int lane  = threadIdx.x & 31;
  int row0  = blockIdx.x * 64 + (wave >> 1) * 16;
  int col0  = (wave & 1) * 64;

  const _Float16* wsel = w16 + which * CH * CH;
  const float*    bias = (which == 0) ? bq : (which == 1) ? bk : bv;
  const _Float16* ta   = t16 + ((size_t)b * SEQ + row0) * CH;

  v16h af[4];
#pragma unroll
  for (int kc = 0; kc < 4; ++kc) af[kc] = load_frag_a(ta + kc * 32, CH, lane);

  v8f acc[4] = {};
#pragma unroll
  for (int ct = 0; ct < 4; ++ct)
#pragma unroll
    for (int kc = 0; kc < 4; ++kc)
      acc[ct] = wmma16(af[kc],
                       load_frag_b(wsel + (col0 + ct * 16) * CH + kc * 32, CH, lane),
                       acc[ct]);

  int rlo = (lane >> 4) * 8;
  int nl  = lane & 15;
#pragma unroll
  for (int ct = 0; ct < 4; ++ct)
#pragma unroll
    for (int r = 0; r < 8; ++r) {
      int row = row0 + r + rlo;
      int col = col0 + ct * 16 + nl;
      float v = acc[ct][r] + bias[col];
      if (which == 0)
        q16[((size_t)b * SEQ + row) * CH + col] = (_Float16)(v * ATT_SCALE);
      else if (which == 1)
        k16[((size_t)b * SEQ + row) * CH + col] = (_Float16)v;
      else
        vT16[((size_t)b * CH + col) * SEQ + row] = (_Float16)v;  // transposed
    }
}

// ---------------------------------------------------------------------------
// Kernel 3: flash attention + fused output projection + residual.
// grid (S/64, B), 4 waves/block; each wave owns a 16-row query strip.
// Online softmax; P re-laid out C/D->A through a private per-wave LDS slice
// (LDS ops are in-order within a wave, no barrier needed).
// ---------------------------------------------------------------------------
__global__ __launch_bounds__(128) void attn_kernel(const _Float16* __restrict__ q16,
                                                   const _Float16* __restrict__ k16,
                                                   const _Float16* __restrict__ vT16,
                                                   const _Float16* __restrict__ wo16,
                                                   const float* __restrict__ bo,
                                                   const float* __restrict__ x,
                                                   float* __restrict__ out) {
  __shared__ _Float16 lds[4][16 * 128];   // per-wave staging (P: 16x64, O: 16x128)
  int b    = blockIdx.y;
  int wave = threadIdx.x >> 5;
  int lane = threadIdx.x & 31;
  int q0   = blockIdx.x * 64 + wave * 16;
  _Float16* pbuf = &lds[wave][0];
  int rlo = (lane >> 4) * 8;
  int nl  = lane & 15;

  // Q fragments (already scaled by 1/sqrt(512))
  v16h qf[4];
  const _Float16* qp = q16 + ((size_t)b * SEQ + q0) * CH;
#pragma unroll
  for (int kc = 0; kc < 4; ++kc) qf[kc] = load_frag_a(qp + kc * 32, CH, lane);

  v8f oacc[8] = {};
  float mst[8], lst[8];
#pragma unroll
  for (int r = 0; r < 8; ++r) { mst[r] = -INFINITY; lst[r] = 0.f; }

  for (int kt = 0; kt < SEQ / 64; ++kt) {
    int key0 = kt * 64;
    // S = Q K^T  (64 keys = 4 col tiles)
    v8f sacc[4] = {};
    const _Float16* kp = k16 + ((size_t)b * SEQ + key0) * CH;
#pragma unroll
    for (int ct = 0; ct < 4; ++ct)
#pragma unroll
      for (int kc = 0; kc < 4; ++kc)
        sacc[ct] = wmma16(qf[kc],
                          load_frag_b(kp + ct * 16 * CH + kc * 32, CH, lane),
                          sacc[ct]);

    // online softmax: row max across the 16 lanes holding each row
    float mnew[8], alpha[8], rs[8];
#pragma unroll
    for (int r = 0; r < 8; ++r) {
      float mx = fmaxf(fmaxf(sacc[0][r], sacc[1][r]), fmaxf(sacc[2][r], sacc[3][r]));
#pragma unroll
      for (int off = 8; off >= 1; off >>= 1) mx = fmaxf(mx, __shfl_xor(mx, off, 32));
      mnew[r]  = fmaxf(mst[r], mx);
      alpha[r] = __expf(mst[r] - mnew[r]);
      rs[r]    = 0.f;
    }
    // P = exp(S - m), stage as f16 row-major 16x64 in this wave's LDS slice
#pragma unroll
    for (int ct = 0; ct < 4; ++ct)
#pragma unroll
      for (int r = 0; r < 8; ++r) {
        float p = __expf(sacc[ct][r] - mnew[r]);
        rs[r] += p;
        pbuf[(r + rlo) * 64 + ct * 16 + nl] = (_Float16)p;
      }
#pragma unroll
    for (int r = 0; r < 8; ++r) {
      float t = rs[r];
#pragma unroll
      for (int off = 8; off >= 1; off >>= 1) t += __shfl_xor(t, off, 32);
      lst[r] = lst[r] * alpha[r] + t;
      mst[r] = mnew[r];
    }
#pragma unroll
    for (int cc = 0; cc < 8; ++cc)
#pragma unroll
      for (int r = 0; r < 8; ++r) oacc[cc][r] *= alpha[r];

    // O += P V  (A-fragments of P come back from LDS; V^T gives contiguous B)
    v16h pa0 = load_frag_a(pbuf, 64, lane);
    v16h pa1 = load_frag_a(pbuf + 32, 64, lane);
#pragma unroll
    for (int cc = 0; cc < 8; ++cc) {
      const _Float16* vp = vT16 + ((size_t)b * CH + cc * 16) * SEQ + key0;
      oacc[cc] = wmma16(pa0, load_frag_b(vp, SEQ, lane), oacc[cc]);
      oacc[cc] = wmma16(pa1, load_frag_b(vp + 32, SEQ, lane), oacc[cc]);
    }
  }

  // normalize and stage O (16x128 f16) for the fused output projection
#pragma unroll
  for (int cc = 0; cc < 8; ++cc)
#pragma unroll
    for (int r = 0; r < 8; ++r)
      pbuf[(r + rlo) * 128 + cc * 16 + nl] = (_Float16)(oacc[cc][r] / lst[r]);

  v16h oa[4];
#pragma unroll
  for (int kc = 0; kc < 4; ++kc) oa[kc] = load_frag_a(pbuf + kc * 32, CH, lane);

  v8f pacc[8] = {};
#pragma unroll
  for (int ct = 0; ct < 8; ++ct)
#pragma unroll
    for (int kc = 0; kc < 4; ++kc)
      pacc[ct] = wmma16(oa[kc],
                        load_frag_b(wo16 + ct * 16 * CH + kc * 32, CH, lane),
                        pacc[ct]);

  // bias + residual, write back in [B, C, H*W] fp32
#pragma unroll
  for (int ct = 0; ct < 8; ++ct)
#pragma unroll
    for (int r = 0; r < 8; ++r) {
      int row = q0 + r + rlo;          // spatial position
      int col = ct * 16 + nl;          // channel
      size_t idx = ((size_t)b * CH + col) * SEQ + row;
      out[idx] = pacc[ct][r] + bo[col] + x[idx];
    }
}

// ---------------------------------------------------------------------------
extern "C" void kernel_launch(void* const* d_in, const int* in_sizes, int n_in,
                              void* d_out, int out_size, void* d_ws, size_t ws_size,
                              hipStream_t stream) {
  const float* x    = (const float*)d_in[0];
  const float* gn_w = (const float*)d_in[1];
  const float* gn_b = (const float*)d_in[2];
  const float* wq   = (const float*)d_in[3];
  const float* bq   = (const float*)d_in[4];
  const float* wk   = (const float*)d_in[5];
  const float* bk   = (const float*)d_in[6];
  const float* wv   = (const float*)d_in[7];
  const float* bv   = (const float*)d_in[8];
  const float* wo   = (const float*)d_in[9];
  const float* bo   = (const float*)d_in[10];

  const size_t TOK = (size_t)BATCH * SEQ * CH;   // 8,388,608 halves = 16 MB
  _Float16* ws   = (_Float16*)d_ws;
  _Float16* t16  = ws;
  _Float16* q16  = ws + TOK;
  _Float16* k16  = ws + 2 * TOK;
  _Float16* vT16 = ws + 3 * TOK;
  _Float16* w16  = ws + 4 * TOK;                 // 4 * 16384 halves

  wconv_kernel<<<(CH * CH + 255) / 256, 256, 0, stream>>>(wq, wk, wv, wo, w16);
  gn_kernel<<<BATCH * NGRP, 256, 0, stream>>>(x, gn_w, gn_b, t16);
  qkv_kernel<<<dim3(SEQ / 64, 3, BATCH), 256, 0, stream>>>(t16, w16, bq, bk, bv,
                                                           q16, k16, vT16);
  attn_kernel<<<dim3(SEQ / 64, BATCH), 128, 0, stream>>>(q16, k16, vT16,
                                                         w16 + 3 * CH * CH, bo,
                                                         x, (float*)d_out);
}
